// HaloAttention_22084721836725
// MI455X (gfx1250) — matrix operand order repaired and checked by reading, not compiled, not run on hardware
//
#include <hip/hip_runtime.h>
#include <cstdint>
#include <cstddef>

// ---------------- constants (match reference) ----------------
#define NHEADS 8
#define DHEAD 32
#define BLK 8
#define HALO 3
#define WIN 14           // BLK + 2*HALO
#define NKEY 196         // WIN*WIN
#define KPAD 224         // 7 * 32 (K padding for P@V)
#define NREL 27          // 2*WIN - 1
#define IMG 96
#define NBH 12           // IMG/BLK
#define NBLK 144         // NBH*NBH
#define HW 9216          // IMG*IMG
#define CIN 256
#define MOUT 768         // 256 q + 512 kv
#define SCALE_F 0.17677669529663687f  // 32^-0.5

typedef __attribute__((ext_vector_type(16))) __bf16 v16bf;
typedef __attribute__((ext_vector_type(8)))  float  v8f;

struct __align__(16) U4 { unsigned int x, y, z, w; };
union Frag { U4 u[2]; v16bf v; };

__device__ inline unsigned short f2bf(float f) {
  union { float f; unsigned int u; } c; c.f = f;
  unsigned int u = c.u;
  return (unsigned short)((u + 0x7FFFu + ((u >> 16) & 1u)) >> 16);
}

__device__ inline v8f zero8() {
  v8f z = {0.f, 0.f, 0.f, 0.f, 0.f, 0.f, 0.f, 0.f};
  return z;
}

// ---------------- phase 1a: weights fp32 -> bf16, fused [768][256] ----------------
__global__ void convert_weights(const float* __restrict__ qw,
                                const float* __restrict__ kvw,
                                unsigned short* __restrict__ wb) {
  int idx = blockIdx.x * 256 + threadIdx.x;   // 768*256 total
  if (idx >= MOUT * CIN) return;
  int o = idx >> 8, c = idx & 255;
  float v = (o < 256) ? qw[o * 256 + c] : kvw[(o - 256) * 256 + c];
  wb[idx] = f2bf(v);
}

// ---------------- phase 1b: x[b][c][hw] fp32 -> xT[b][hw][c] bf16 ----------------
__global__ void transpose_x(const float* __restrict__ x,
                            unsigned short* __restrict__ xT) {
  __shared__ float t[32][33];
  int b = blockIdx.z, c0 = blockIdx.y * 32, hw0 = blockIdx.x * 32;
  int tx = threadIdx.x, ty = threadIdx.y;
  const float* xb = x + (size_t)b * CIN * HW;
#pragma unroll
  for (int i = 0; i < 4; i++) {
    int c = c0 + ty + i * 8;
    t[ty + i * 8][tx] = xb[(size_t)c * HW + hw0 + tx];
  }
  __syncthreads();
  unsigned short* dst = xT + (size_t)b * HW * CIN;
#pragma unroll
  for (int i = 0; i < 4; i++) {
    int hw = hw0 + ty + i * 8;
    dst[(size_t)hw * CIN + c0 + tx] = f2bf(t[tx][ty + i * 8]);
  }
}

// ---------------- phase 2: QKV projection GEMM via WMMA ----------------
// D[o][hw] = W[o][c] * xT[hw][c]^T, M=768, N=9216 per batch, K=256 (8 wmma steps)
// grid: (144 N-groups of 64, 12 M-groups of 64, 8 batches); block: 128 (4 waves)
__global__ __launch_bounds__(128) void qkv_gemm(const unsigned short* __restrict__ wb,
                                                const unsigned short* __restrict__ xT,
                                                unsigned short* __restrict__ qg,
                                                unsigned short* __restrict__ kg,
                                                unsigned short* __restrict__ vg) {
  int b    = blockIdx.z;
  int wave = threadIdx.x >> 5;
  int lane = threadIdx.x & 31;
  int lo   = lane & 15;
  int hi   = lane >> 4;

  int mtile = blockIdx.y * 4 + wave;          // 16-row tile of o
  int n0    = blockIdx.x * 64;                // 64 hw columns

  const unsigned short* Arow = wb + (size_t)(mtile * 16 + lo) * CIN;
  const unsigned short* xb   = xT + (size_t)b * HW * CIN;
  int aklo   = hi * 8;    // A fragment: chunks at k=aklo and k=16+aklo
  int bkbase = hi * 8;    // B fragment: chunks at k=bkbase and k=16+bkbase

  v8f acc[4] = { zero8(), zero8(), zero8(), zero8() };

#pragma unroll
  for (int ks = 0; ks < 8; ks++) {
    int k0 = ks * 32;
    Frag a;
    a.u[0] = *(const U4*)(Arow + k0 + aklo);
    a.u[1] = *(const U4*)(Arow + k0 + 16 + aklo);
#pragma unroll
    for (int nt = 0; nt < 4; nt++) {
      int hw = n0 + nt * 16 + lo;
      const unsigned short* Brow = xb + (size_t)hw * CIN + k0 + bkbase;
      Frag bf_;
      bf_.u[0] = *(const U4*)(Brow);
      bf_.u[1] = *(const U4*)(Brow + 16);
      acc[nt] = __builtin_amdgcn_wmma_f32_16x16x32_bf16(
          false, a.v, false, bf_.v, (short)0, acc[nt], false, false);
    }
  }

  // C/D layout: vgpr r, lanes 0-15 -> row r, lanes 16-31 -> row r+8, col = lane&15
  int obase = mtile * 16 + hi * 8;   // this lane holds o = obase..obase+7
#pragma unroll
  for (int nt = 0; nt < 4; nt++) {
    int hw = n0 + nt * 16 + lo;
    union { unsigned short s[8]; U4 u; } ov;
#pragma unroll
    for (int r = 0; r < 8; r++) ov.s[r] = f2bf(acc[nt][r]);
    unsigned short* dst;
    if (obase < 256) {                         // Q channels
      int head = obase >> 5, d = obase & 31;
      dst = qg + (((size_t)(b * 8 + head) * HW + hw) * DHEAD + d);
    } else {                                   // KV channels (64 per head)
      int oo = obase - 256;
      int head = oo >> 6, within = oo & 63;
      if (within < 32)
        dst = kg + (((size_t)(b * 8 + head) * HW + hw) * DHEAD + within);
      else
        dst = vg + (((size_t)(b * 8 + head) * HW + hw) * DHEAD + (within - 32));
    }
    *(U4*)dst = ov.u;
  }
}

// ---------------- phase 3: halo attention, one workgroup per (bh, block) ----------------
// grid: (144 blocks, 64 bh); block: 128 threads (4 waves, wave w owns query rows [16w,16w+16))
__global__ __launch_bounds__(128) void halo_attn(const unsigned short* __restrict__ qg,
                                                 const unsigned short* __restrict__ kg,
                                                 const unsigned short* __restrict__ vg,
                                                 const float* __restrict__ hrel,
                                                 const float* __restrict__ wrel,
                                                 float* __restrict__ out) {
  // kv_s: K' matrix [256 rows][32 d] (16KB), later reused as V^T [32][224] (14KB)
  __shared__ unsigned short kv_s[256 * 32];
  __shared__ float relS[64][56];               // per-row q.height_rel[0..26], q.width_rel[27..53]
  __shared__ unsigned short pLds[64][KPAD];    // softmax probs, bf16

  int nb = blockIdx.x;
  int br = nb / NBH, bwc = nb % NBH;
  int bh = blockIdx.y;
  int b = bh >> 3, head = bh & 7;
  int tid = threadIdx.x;
  int wave = tid >> 5, lane = tid & 31, lo = lane & 15, hi = lane >> 4;

  // ---- fill K' LDS: rows 0..195 = halo window keys, 196..222 = height_rel,
  //      223..249 = width_rel, 250..255 = zero
  {
    const unsigned short* kbh = kg + (size_t)bh * HW * DHEAD;
    for (int row = tid * 2; row < tid * 2 + 2; row++) {
      unsigned short* dstrow = kv_s + row * DHEAD;
      if (row < NKEY) {
        int i = row / WIN, j = row % WIN;
        int hs = br * BLK - HALO + i, ws = bwc * BLK - HALO + j;
        if (hs >= 0 && hs < IMG && ws >= 0 && ws < IMG) {
          const U4* src = (const U4*)(kbh + ((size_t)hs * IMG + ws) * DHEAD);
          ((U4*)dstrow)[0] = src[0];
          ((U4*)dstrow)[1] = src[1];
        } else {
          for (int d = 0; d < DHEAD; d++) dstrow[d] = 0;
        }
      } else if (row < NKEY + NREL) {
        const float* rv = hrel + (row - NKEY) * DHEAD;
        for (int d = 0; d < DHEAD; d++) dstrow[d] = f2bf(rv[d]);
      } else if (row < NKEY + 2 * NREL) {
        const float* rv = wrel + (row - (NKEY + NREL)) * DHEAD;
        for (int d = 0; d < DHEAD; d++) dstrow[d] = f2bf(rv[d]);
      } else {
        for (int d = 0; d < DHEAD; d++) dstrow[d] = 0;
      }
    }
  }
  __syncthreads();

  // ---- load Q A-fragment straight from global (rows wave*16..+16, K=32)
  int qrow = wave * 16 + lo;                      // query index in block
  int hb = qrow >> 3, wb2 = qrow & 7;
  const unsigned short* qrowp =
      qg + (((size_t)bh * HW) + (br * BLK + hb) * IMG + (bwc * BLK + wb2)) * DHEAD;
  int aklo = hi * 8;
  int bkbase = hi * 8;
  Frag aq;
  aq.u[0] = *(const U4*)(qrowp + aklo);
  aq.u[1] = *(const U4*)(qrowp + 16 + aklo);

  // ---- S = Q K'^T : one K-step (K=32), 16 N-tiles of 16 columns
  v8f acc[16];
#pragma unroll
  for (int t = 0; t < 16; t++) {
    int key = t * 16 + lo;
    const unsigned short* krow = kv_s + key * DHEAD + bkbase;
    Frag bf_;
    bf_.u[0] = *(const U4*)(krow);
    bf_.u[1] = *(const U4*)(krow + 16);
    acc[t] = __builtin_amdgcn_wmma_f32_16x16x32_bf16(
        false, aq.v, false, bf_.v, (short)0, zero8(), false, false);
  }

  // ---- spill rel columns (196..249) to relS (only this wave's rows; no barrier needed)
#pragma unroll
  for (int t = 12; t < 16; t++) {
    int n = t * 16 + lo;
#pragma unroll
    for (int r = 0; r < 8; r++) {
      int m = wave * 16 + hi * 8 + r;
      if (n >= NKEY && n < NKEY + NREL)            relS[m][n - NKEY] = acc[t][r];
      else if (n >= NKEY + NREL && n < NKEY + 2 * NREL) relS[m][NREL + n - (NKEY + NREL)] = acc[t][r];
    }
  }

  // ---- logits = S*scale + rel_h + rel_w, track row max (overwrite acc in place)
  float rmax[8];
#pragma unroll
  for (int r = 0; r < 8; r++) rmax[r] = -1e30f;
#pragma unroll
  for (int t = 0; t < 13; t++) {
    int n = t * 16 + lo;
    int ki = n / WIN, kj = n - ki * WIN;          // only meaningful for n < 196
#pragma unroll
    for (int r = 0; r < 8; r++) {
      int m = wave * 16 + hi * 8 + r;
      int h = m >> 3, w = m & 7;
      float lg;
      if (n < NKEY)
        lg = acc[t][r] * SCALE_F + relS[m][ki - h + 13] + relS[m][NREL + kj - w + 13];
      else
        lg = -1e30f;
      acc[t][r] = lg;
      rmax[r] = fmaxf(rmax[r], lg);
    }
  }
#pragma unroll
  for (int r = 0; r < 8; r++)
#pragma unroll
    for (int mask = 1; mask < 16; mask <<= 1)
      rmax[r] = fmaxf(rmax[r], __shfl_xor(rmax[r], mask, 32));

  // ---- exp and row sum
  float rsum[8];
#pragma unroll
  for (int r = 0; r < 8; r++) rsum[r] = 0.f;
#pragma unroll
  for (int t = 0; t < 13; t++) {
#pragma unroll
    for (int r = 0; r < 8; r++) {
      float e = __expf(acc[t][r] - rmax[r]);      // padded cols held -1e30 -> exp == 0
      acc[t][r] = e;
      rsum[r] += e;
    }
  }
#pragma unroll
  for (int r = 0; r < 8; r++)
#pragma unroll
    for (int mask = 1; mask < 16; mask <<= 1)
      rsum[r] += __shfl_xor(rsum[r], mask, 32);
  float rinv[8];
#pragma unroll
  for (int r = 0; r < 8; r++) rinv[r] = 1.0f / rsum[r];

  // ---- write P (bf16) to LDS, cols 0..223 (196..223 forced to zero)
#pragma unroll
  for (int t = 0; t < 14; t++) {
    int n = t * 16 + lo;
#pragma unroll
    for (int r = 0; r < 8; r++) {
      int m = wave * 16 + hi * 8 + r;
      unsigned short pv = 0;
      if (t < 13 && n < NKEY) pv = f2bf(acc[t][r] * rinv[r]);
      pLds[m][n] = pv;
    }
  }
  __syncthreads();   // everyone done reading K' from kv_s before it is reused for V^T

  // ---- load V^T into kv_s: vT[d][key], key 196..223 zero
  {
    const unsigned short* vbh = vg + (size_t)bh * HW * DHEAD;
    for (int key = tid; key < KPAD; key += 128) {
      bool ok = false;
      const unsigned short* src = nullptr;
      if (key < NKEY) {
        int i = key / WIN, j = key % WIN;
        int hs = br * BLK - HALO + i, ws = bwc * BLK - HALO + j;
        if (hs >= 0 && hs < IMG && ws >= 0 && ws < IMG) {
          ok = true;
          src = vbh + ((size_t)hs * IMG + ws) * DHEAD;
        }
      }
      if (ok) {
        for (int d = 0; d < DHEAD; d++) kv_s[d * KPAD + key] = src[d];
      } else {
        for (int d = 0; d < DHEAD; d++) kv_s[d * KPAD + key] = 0;
      }
    }
  }
  __syncthreads();

  // ---- O = P V : M=16 (wave rows), N=32 (2 tiles), K=224 (7 steps)
  v8f oacc[2] = { zero8(), zero8() };
  int prow = wave * 16 + lo;
#pragma unroll
  for (int kk = 0; kk < 7; kk++) {
    Frag pa;
    pa.u[0] = *(const U4*)(&pLds[prow][kk * 32 + aklo]);
    pa.u[1] = *(const U4*)(&pLds[prow][kk * 32 + 16 + aklo]);
#pragma unroll
    for (int nt = 0; nt < 2; nt++) {
      int d = nt * 16 + lo;
      Frag vb;
      vb.u[0] = *(const U4*)(kv_s + d * KPAD + kk * 32 + bkbase);
      vb.u[1] = *(const U4*)(kv_s + d * KPAD + kk * 32 + bkbase + 16);
      oacc[nt] = __builtin_amdgcn_wmma_f32_16x16x32_bf16(
          false, pa.v, false, vb.v, (short)0, oacc[nt], false, false);
    }
  }

  // ---- store: out[b][head*32+d][br*8 + h_in][bw*8 + 0..7], 8 consecutive w per lane
  int h_in = 2 * wave + hi;
#pragma unroll
  for (int nt = 0; nt < 2; nt++) {
    int d = nt * 16 + lo;
    float* dst = out + ((((size_t)b * 256 + head * 32 + d) * IMG + br * BLK + h_in) * IMG
                        + bwc * BLK);
    struct __align__(16) F4 { float a, bb, c, dd; };
    F4 v0 = { oacc[nt][0], oacc[nt][1], oacc[nt][2], oacc[nt][3] };
    F4 v1 = { oacc[nt][4], oacc[nt][5], oacc[nt][6], oacc[nt][7] };
    *(F4*)(dst)     = v0;
    *(F4*)(dst + 4) = v1;
  }
}

// ---------------- launch ----------------
extern "C" void kernel_launch(void* const* d_in, const int* in_sizes, int n_in,
                              void* d_out, int out_size, void* d_ws, size_t ws_size,
                              hipStream_t stream) {
  (void)in_sizes; (void)n_in; (void)out_size; (void)ws_size;
  const float* x    = (const float*)d_in[0];
  const float* qw   = (const float*)d_in[1];
  const float* kvw  = (const float*)d_in[2];
  const float* hrel = (const float*)d_in[3];
  const float* wrel = (const float*)d_in[4];
  float* out = (float*)d_out;

  char* ws = (char*)d_ws;
  size_t off = 0;
  unsigned short* wb = (unsigned short*)(ws + off);
  off += (size_t)MOUT * CIN * 2;
  off = (off + 255) & ~(size_t)255;
  unsigned short* xT = (unsigned short*)(ws + off);
  off += (size_t)8 * HW * CIN * 2;
  unsigned short* qg = (unsigned short*)(ws + off);
  off += (size_t)64 * HW * DHEAD * 2;
  unsigned short* kg = (unsigned short*)(ws + off);
  off += (size_t)64 * HW * DHEAD * 2;
  unsigned short* vg = (unsigned short*)(ws + off);
  off += (size_t)64 * HW * DHEAD * 2;   // total ~144.5 MB of workspace

  convert_weights<<<768, 256, 0, stream>>>(qw, kvw, wb);
  transpose_x<<<dim3(HW / 32, CIN / 32, 8), dim3(32, 8), 0, stream>>>(x, xT);
  qkv_gemm<<<dim3(HW / 64, MOUT / 64, 8), 128, 0, stream>>>(wb, xT, qg, kg, vg);
  halo_attn<<<dim3(NBLK, 64), 128, 0, stream>>>(qg, kg, vg, hrel, wrel, out);
}
